// MFA_Block_48808008352102
// MI455X (gfx1250) — matrix operand order repaired and checked by reading, not compiled, not run on hardware
//
#include <hip/hip_runtime.h>
#include <hip/hip_bf16.h>
#include <stdint.h>

typedef __bf16 bf16;
typedef __attribute__((ext_vector_type(16))) __bf16 v16bf;
typedef __attribute__((ext_vector_type(8)))  __bf16 v8bf;
typedef __attribute__((ext_vector_type(8)))  float  v8f;

typedef unsigned int u32;
typedef __attribute__((ext_vector_type(4))) u32 v4u32;
typedef __attribute__((ext_vector_type(8))) int  v8i32;
typedef __attribute__((ext_vector_type(4))) int  v4i32;

#define BN_TOTAL 32768L   // B*N = 8*4096
#define HC 768
#define NL_TOK 4096
#define NBATCH 8
#define BN_EPS 1e-5f

#if defined(__has_builtin)
#  if __has_builtin(__builtin_amdgcn_tensor_load_to_lds)
#    define HAVE_TDM 1
#  endif
#endif
#ifndef HAVE_TDM
#  define HAVE_TDM 0
#endif

// ---------------------------------------------------------------------------
// TDM: issue one 2D tile load (rows x cols bf16, row pitch in elements) into
// LDS at byte offset lds_addr. Descriptor packed per CDNA5 D# spec:
//   group0: [1:0]count=1 | lds_addr | global_addr[56:0] | type=2
//   group1: data_size=1(2B) | tensor_dim0/1 | tile_dim0/1 | dim0_stride
// Groups 2/3 zero (2D tile: tile_dim2..4 = 0).
// ---------------------------------------------------------------------------
#if HAVE_TDM
__device__ __forceinline__
void tdm_load_2d(u32 lds_addr, const void* gptr, u32 rows, u32 cols, u32 pitch) {
  uint64_t ga = (uint64_t)(uintptr_t)gptr;
  v4u32 g0;
  g0[0] = 1u;                                   // count=1, user descriptor
  g0[1] = lds_addr;                             // LDS byte address
  g0[2] = (u32)ga;                              // global addr [31:0]
  g0[3] = (u32)((ga >> 32) & 0x01FFFFFFu) | (2u << 30);  // addr[56:32] | type=2
  v8i32 g1;
  g1[0] = (int)(1u << 16);                      // workgroup_mask=0, data_size=1 (2B)
  g1[1] = (int)((cols & 0xFFFFu) << 16);        // tensor_dim0 lo16 (abar addr = 0)
  g1[2] = (int)(((cols >> 16) & 0xFFFFu) | ((rows & 0xFFFFu) << 16)); // d0 hi | d1 lo
  g1[3] = (int)(((rows >> 16) & 0xFFFFu) | ((cols & 0xFFFFu) << 16)); // d1 hi | tile_dim0
  g1[4] = (int)(rows & 0xFFFFu);                // tile_dim1 (tile_dim2 = 0)
  g1[5] = (int)pitch;                           // tensor_dim0_stride lo32
  g1[6] = 0;                                    // stride hi16 | dim1_stride lo
  g1[7] = 0;
  v4i32 zz = {0, 0, 0, 0};
#if (__clang_major__ >= 23)
  v8i32 z8 = {0, 0, 0, 0, 0, 0, 0, 0};
  __builtin_amdgcn_tensor_load_to_lds(g0, g1, zz, zz, z8, 0);
#else
  __builtin_amdgcn_tensor_load_to_lds(g0, g1, zz, zz, 0);
#endif
}
#endif

// ---------------------------------------------------------------------------
// f32 -> bf16 cast / zero (grid-stride)
// ---------------------------------------------------------------------------
__global__ __launch_bounds__(256)
void cast_f32_to_bf16(const float* __restrict__ in, bf16* __restrict__ out, long n) {
  long i = (long)blockIdx.x * blockDim.x + threadIdx.x;
  long stride = (long)gridDim.x * blockDim.x;
  for (; i < n; i += stride) out[i] = (bf16)in[i];
}

__global__ __launch_bounds__(256)
void zero_f32(float* __restrict__ p, long n) {
  long i = (long)blockIdx.x * blockDim.x + threadIdx.x;
  long stride = (long)gridDim.x * blockDim.x;
  for (; i < n; i += stride) p[i] = 0.0f;
}

// ---------------------------------------------------------------------------
// Unified bf16 WMMA GEMM:  C[m,n] = scale * sum_k A[m,k]*W[n,k]  (+ bias[n])
// 128 threads = 4 waves, 64x64 block tile, 32x32 per wave.
// Double-buffered LDS pipeline: wave 0 issues TDM tile loads (A 64x32,
// B 64x32) for step kt+1, waits s_wait_tensorcnt<=2 for step kt's pair,
// then all waves consume fragments from LDS via ds_load_b128.
// ---------------------------------------------------------------------------
__global__ __launch_bounds__(128)
void wmma_gemm(const bf16* __restrict__ A, long a_rs, long a_bo,
               const bf16* __restrict__ W, long b_rs, long b_bo,
               bf16* __restrict__ Cb, float* __restrict__ Cf,
               int c_mode, long c_rs, long c_bo,
               const float* __restrict__ bias, float scale,
               int M, int N, int K)
{
  __shared__ __align__(16) bf16 smA[2][64 * 32];
  __shared__ __align__(16) bf16 smB[2][64 * 32];

  const int tid  = threadIdx.x;
  const int lane = tid & 31;
  const int wave = tid >> 5;        // 0..3
  const int wm   = wave >> 1;       // 0..1 (M dir)
  const int wn   = wave & 1;        // 0..1 (N dir)
  const int l15  = lane & 15;
  const int hi   = lane >> 4;       // 0 or 1
  const long z   = blockIdx.z;

  const bf16* Ab = A + z * a_bo + (long)blockIdx.y * 64 * a_rs;
  const bf16* Bb = W + z * b_bo + (long)blockIdx.x * 64 * b_rs;

  const int nK = K >> 5;

  // ---- tile staging (TDM if available, cooperative copy otherwise) ----
#if HAVE_TDM
#define STAGE(kt, buf)                                                         \
  do {                                                                         \
    if (tid < 32) {                                                            \
      tdm_load_2d((u32)(uintptr_t)&smA[(buf)][0], Ab + (long)(kt) * 32,        \
                  64u, 32u, (u32)a_rs);                                        \
      tdm_load_2d((u32)(uintptr_t)&smB[(buf)][0], Bb + (long)(kt) * 32,        \
                  64u, 32u, (u32)b_rs);                                        \
    }                                                                          \
  } while (0)
#else
#define STAGE(kt, buf)                                                         \
  do {                                                                         \
    const int rw = tid >> 1, hf = (tid & 1) * 16;                              \
    const bf16* gA = Ab + (long)rw * a_rs + (long)(kt) * 32 + hf;              \
    const bf16* gB = Bb + (long)rw * b_rs + (long)(kt) * 32 + hf;              \
    *(v8bf*)&smA[(buf)][rw * 32 + hf]     = *(const v8bf*)gA;                  \
    *(v8bf*)&smA[(buf)][rw * 32 + hf + 8] = *(const v8bf*)(gA + 8);            \
    *(v8bf*)&smB[(buf)][rw * 32 + hf]     = *(const v8bf*)gB;                  \
    *(v8bf*)&smB[(buf)][rw * 32 + hf + 8] = *(const v8bf*)(gB + 8);            \
  } while (0)
#endif

  // per-lane K sub-offsets from the 16-bit WMMA fragment layouts:
  //   A (16x32): lanes 0-15 hold K {0..7,16..23}, lanes 16-31 K {8..15,24..31}
  //   B (32x16): lanes 0-15 hold K {0..7,16..23}, lanes 16-31 K {8..15,24..31}
  const int ka = hi * 8;
  const int kb = hi * 8;

  v8f acc[2][2] = {};

  STAGE(0, 0);                       // prologue
  for (int kt = 0; kt < nK; ++kt) {
    const int buf = kt & 1;
    const bool hasNext = (kt + 1) < nK;
    if (hasNext) STAGE(kt + 1, buf ^ 1);
#if HAVE_TDM
    if (tid < 32) {
      if (hasNext) __builtin_amdgcn_s_wait_tensorcnt(2);  // current pair done
      else         __builtin_amdgcn_s_wait_tensorcnt(0);
    }
#endif
    __syncthreads();

    v16bf afrag[2], bfrag[2];
#pragma unroll
    for (int i = 0; i < 2; ++i) {
      const bf16* ap = &smA[buf][(wm * 32 + i * 16 + l15) * 32 + ka];
      union { v16bf v; v8bf h[2]; } ua;
      ua.h[0] = *(const v8bf*)(ap);
      ua.h[1] = *(const v8bf*)(ap + 16);
      afrag[i] = ua.v;
      const bf16* bp = &smB[buf][(wn * 32 + i * 16 + l15) * 32 + kb];
      union { v16bf v; v8bf h[2]; } ub;
      ub.h[0] = *(const v8bf*)(bp);
      ub.h[1] = *(const v8bf*)(bp + 16);
      bfrag[i] = ub.v;
    }
#pragma unroll
    for (int i = 0; i < 2; ++i)
#pragma unroll
      for (int j = 0; j < 2; ++j)
        acc[i][j] = __builtin_amdgcn_wmma_f32_16x16x32_bf16(
            false, afrag[i], false, bfrag[j], (short)0, acc[i][j], false, false);

    __syncthreads();                 // protect buf before it is re-staged
  }

  // C/D fragment: VGPR v -> row m = v + (lane>=16 ? 8 : 0), col = lane&15
  const int mTile = blockIdx.y * 64 + wm * 32;
  const int nTile = blockIdx.x * 64 + wn * 32;
#pragma unroll
  for (int i = 0; i < 2; ++i) {
#pragma unroll
    for (int j = 0; j < 2; ++j) {
      const int mBase = mTile + i * 16 + hi * 8;
      const int nCol  = nTile + j * 16 + l15;
      const float bv  = bias ? bias[nCol] : 0.0f;
#pragma unroll
      for (int v = 0; v < 8; ++v) {
        const long m  = mBase + v;
        float val = acc[i][j][v] * scale + bv;
        if (c_mode == 2)      Cf[z * c_bo + m * c_rs + nCol] = val;
        else if (c_mode == 1) Cb[z * c_bo + (long)nCol * c_rs + m] = (bf16)val;
        else                  Cb[z * c_bo + m * c_rs + nCol] = (bf16)val;
      }
    }
  }
#undef STAGE
}

// ---------------------------------------------------------------------------
// BatchNorm stats: per-channel sum & sum-of-squares over all B*N rows.
// C = 768 = 3*256; each thread owns 3 channels; 128 blocks x 256 rows.
// ---------------------------------------------------------------------------
__global__ __launch_bounds__(256)
void bn_stats(const float* __restrict__ ypre, float* __restrict__ stats) {
  const int c0 = threadIdx.x, c1 = c0 + 256, c2 = c0 + 512;
  float s0 = 0, s1 = 0, s2 = 0, q0 = 0, q1 = 0, q2 = 0;
  const long base = (long)blockIdx.x * 256;
  for (int r = 0; r < 256; ++r) {
    const float* row = ypre + (base + r) * (long)HC;
    float v0 = row[c0], v1 = row[c1], v2 = row[c2];
    s0 += v0; q0 += v0 * v0;
    s1 += v1; q1 += v1 * v1;
    s2 += v2; q2 += v2 * v2;
  }
  atomicAdd(&stats[c0], s0); atomicAdd(&stats[HC + c0], q0);
  atomicAdd(&stats[c1], s1); atomicAdd(&stats[HC + c1], q1);
  atomicAdd(&stats[c2], s2); atomicAdd(&stats[HC + c2], q2);
}

// ---------------------------------------------------------------------------
// BN apply + affine + residual:  out = (y-mu)*rsqrt(var+eps)*gamma + beta + res
// ---------------------------------------------------------------------------
__global__ __launch_bounds__(256)
void bn_apply(const float* __restrict__ ypre, const float* __restrict__ res,
              const float* __restrict__ stats, const float* __restrict__ gamma,
              const float* __restrict__ beta, float* __restrict__ out,
              long total, float invCount) {
  long i = (long)blockIdx.x * blockDim.x + threadIdx.x;
  long stride = (long)gridDim.x * blockDim.x;
  for (; i < total; i += stride) {
    int c = (int)(i % HC);
    float mu  = stats[c] * invCount;
    float var = stats[HC + c] * invCount - mu * mu;
    float r   = rsqrtf(var + BN_EPS);
    out[i] = (ypre[i] - mu) * r * gamma[c] + beta[c] + res[i];
  }
}

// ---------------------------------------------------------------------------
// Host-side orchestration
// ---------------------------------------------------------------------------
static void launch_gemm(hipStream_t s,
                        const bf16* A, long ars, long abo,
                        const bf16* W, long brs, long bbo,
                        bf16* Cb, float* Cf, int cmode, long crs, long cbo,
                        const float* bias, float scale,
                        int M, int N, int K, int Z) {
  dim3 grid(N / 64, M / 64, Z), block(128);
  wmma_gemm<<<grid, block, 0, s>>>(A, ars, abo, W, brs, bbo,
                                   Cb, Cf, cmode, crs, cbo, bias, scale, M, N, K);
}

extern "C" void kernel_launch(void* const* d_in, const int* in_sizes, int n_in,
                              void* d_out, int out_size, void* d_ws, size_t ws_size,
                              hipStream_t stream) {
  const float* x     = (const float*)d_in[0];
  const float* x0    = (const float*)d_in[1];
  const float* c_wg  = (const float*)d_in[2];  const float* c_bg = (const float*)d_in[3];
  const float* c_wt  = (const float*)d_in[4];  const float* c_bt = (const float*)d_in[5];
  const float* c_wp  = (const float*)d_in[6];  const float* c_bp = (const float*)d_in[7];
  const float* c_ww  = (const float*)d_in[8];  const float* c_bw = (const float*)d_in[9];
  const float* c_gamma = (const float*)d_in[10]; const float* c_beta = (const float*)d_in[11];
  const float* p_wg  = (const float*)d_in[12]; const float* p_bg = (const float*)d_in[13];
  const float* p_wt  = (const float*)d_in[14]; const float* p_bt = (const float*)d_in[15];
  const float* p_wp  = (const float*)d_in[16]; const float* p_bp = (const float*)d_in[17];
  const float* p_ww  = (const float*)d_in[18]; const float* p_bw = (const float*)d_in[19];
  const float* p_gamma = (const float*)d_in[20]; const float* p_beta = (const float*)d_in[21];
  (void)in_sizes; (void)n_in; (void)out_size; (void)ws_size;

  // ---- workspace carve-out (256B aligned) ----
  char* wsp = (char*)d_ws;
  auto carve = [&](size_t bytes) {
    char* p = wsp;
    wsp += (bytes + 255) & ~(size_t)255;
    return p;
  };
  bf16*  hb   = (bf16*)carve(BN_TOTAL * HC * 2);      // x_h in bf16 (x, later z)
  bf16*  x0b  = (bf16*)carve(BN_TOTAL * HC * 2);      // x0 in bf16
  bf16*  wgb  = (bf16*)carve((size_t)384 * HC * 2);
  bf16*  wtb  = (bf16*)carve((size_t)384 * HC * 2);
  bf16*  wpb  = (bf16*)carve((size_t)384 * HC * 2);
  bf16*  wwb  = (bf16*)carve((size_t)HC * 384 * 2);
  bf16*  g_t  = (bf16*)carve((size_t)384 * BN_TOTAL * 2);  // g transposed [D][BN]
  bf16*  p_t  = (bf16*)carve((size_t)384 * BN_TOTAL * 2);  // ph transposed [D][BN]
  bf16*  t_bf = (bf16*)carve(BN_TOTAL * 384 * 2);          // th [BN][D]
  bf16*  y_bf = (bf16*)carve(BN_TOTAL * 384 * 2);          // y  [BN][D]
  bf16*  kvt  = (bf16*)carve((size_t)NBATCH * 384 * 384 * 2); // kv^T per batch
  float* ypre = (float*)carve(BN_TOTAL * HC * 4);          // pre-BN output
  float* zbuf = (float*)carve(BN_TOTAL * HC * 4);          // CNL block output
  float* stats = (float*)carve(2 * HC * 4);

  const dim3 cb(256);
  auto cast = [&](const float* in, bf16* out, long n) {
    cast_f32_to_bf16<<<dim3(1024), cb, 0, stream>>>(in, out, n);
  };

  // one non-local block (weights already cast into wgb/wtb/wpb/wwb)
  auto nl_block = [&](int D,
                      const float* bg, const float* bt, const float* bp,
                      const float* bw, const float* gamma, const float* beta,
                      const float* resid, float* outp) {
    const float invN = 1.0f / (float)NL_TOK;
    // th = x_h @ wt^T + bt          -> [BN, D] normal
    launch_gemm(stream, hb, HC, 0, wtb, HC, 0,
                t_bf, nullptr, 0, D, 0, bt, 1.0f, (int)BN_TOTAL, D, HC, 1);
    // g  = x0 @ wg^T + bg           -> transposed [D][BN]
    launch_gemm(stream, x0b, HC, 0, wgb, HC, 0,
                g_t, nullptr, 1, BN_TOTAL, 0, bg, 1.0f, (int)BN_TOTAL, D, HC, 1);
    // ph = x0 @ wp^T + bp           -> transposed [D][BN]
    launch_gemm(stream, x0b, HC, 0, wpb, HC, 0,
                p_t, nullptr, 1, BN_TOTAL, 0, bp, 1.0f, (int)BN_TOTAL, D, HC, 1);
    // kv^T[b][e][d] = (1/N) sum_n g[b,n,e] * ph[b,n,d]   (K = 4096 per batch)
    launch_gemm(stream, g_t, BN_TOTAL, NL_TOK, p_t, BN_TOTAL, NL_TOK,
                kvt, nullptr, 0, D, (long)D * D, nullptr, invN, D, D, NL_TOK, NBATCH);
    // y[b,n,e] = sum_d th[b,n,d] * kv[d,e]   (B rows = kv^T rows, contiguous)
    launch_gemm(stream, t_bf, D, (long)NL_TOK * D, kvt, D, (long)D * D,
                y_bf, nullptr, 0, D, (long)NL_TOK * D, nullptr, 1.0f, NL_TOK, D, D, NBATCH);
    // ypre = y @ ww^T + bw  (fp32 out for BN)
    launch_gemm(stream, y_bf, D, 0, wwb, D, 0,
                nullptr, ypre, 2, HC, 0, bw, 1.0f, (int)BN_TOTAL, HC, D, 1);
    // BatchNorm (training stats) + affine + residual
    zero_f32<<<dim3(8), cb, 0, stream>>>(stats, 2 * HC);
    bn_stats<<<dim3(128), cb, 0, stream>>>(ypre, stats);
    bn_apply<<<dim3(4096), cb, 0, stream>>>(ypre, resid, stats, gamma, beta,
                                            outp, BN_TOTAL * HC, 1.0f / (float)BN_TOTAL);
  };

  // ---- CNL (D = 384), x_h = x ----
  cast(x,  hb,  BN_TOTAL * HC);
  cast(x0, x0b, BN_TOTAL * HC);
  cast(c_wg, wgb, 384L * HC);
  cast(c_wt, wtb, 384L * HC);
  cast(c_wp, wpb, 384L * HC);
  cast(c_ww, wwb, (long)HC * 384);
  nl_block(384, c_bg, c_bt, c_bp, c_bw, c_gamma, c_beta, x, zbuf);

  // ---- PNL (D = 192), x_h = z ----
  cast(zbuf, hb, BN_TOTAL * HC);
  cast(p_wg, wgb, 192L * HC);
  cast(p_wt, wtb, 192L * HC);
  cast(p_wp, wpb, 192L * HC);
  cast(p_ww, wwb, (long)HC * 192);
  nl_block(192, p_bg, p_bt, p_bp, p_bw, p_gamma, p_beta, zbuf, (float*)d_out);
}
